// BitLinear_25340307047079
// MI455X (gfx1250) — compile-verified, matrix operand order checked by reading, not run on hardware
//
#include <hip/hip_runtime.h>
#include <stdint.h>

// ---------------------------------------------------------------------------
// BitLinear forward for MI455X (gfx1250, wave32, WMMA + Tensor Data Mover).
// out = (x @ ternary(w)^T) * gamma + bias, gamma = max(mean|clip(w,±2)|,1e-4)
// Pipeline: gamma reduction -> w: f32->ternary bf16 -> x: f32->bf16 ->
//           bf16 WMMA GEMM (f32 accum); global->LDS tiles moved by TDM
//           (tensor_load_to_lds, 3-stage TENSORcnt pipeline), gamma & bias
//           fused in the epilogue with non-temporal output stores.
// ---------------------------------------------------------------------------

typedef __bf16 bf16;
typedef __attribute__((ext_vector_type(8)))  bf16           v8bf;
typedef __attribute__((ext_vector_type(16))) bf16           v16bf;
typedef __attribute__((ext_vector_type(8)))  float          v8f;
typedef __attribute__((ext_vector_type(4)))  float          f32x4;
typedef __attribute__((ext_vector_type(4)))  unsigned short u16x4;

typedef __attribute__((ext_vector_type(4))) unsigned int tdm_g0_t;  // D# group0
typedef __attribute__((ext_vector_type(8))) int          tdm_g1_t;  // D# group1
typedef __attribute__((ext_vector_type(4))) int          tdm_g2_t;  // D# group2/3
typedef __attribute__((ext_vector_type(8))) int          tdm_g8_t;

static constexpr int MTOT = 8192;    // B*S = 4*2048
static constexpr int NTOT = 16384;   // D_OUT
static constexpr int KTOT = 4096;    // D_IN

__device__ __forceinline__ unsigned short f32_to_bf16_rne(float f) {
    unsigned int u = __builtin_bit_cast(unsigned int, f);
    unsigned int r = u + 0x7fffu + ((u >> 16) & 1u);   // round-to-nearest-even
    return (unsigned short)(r >> 16);
}

// ---- kernel 1: per-block partial sums of |clip(w,-2,2)| (fixed-order) ------
__global__ __launch_bounds__(256) void k_abs_partial(const float* __restrict__ w,
                                                     float* __restrict__ partial) {
    __shared__ float red[256];
    const size_t nchunk = (size_t)NTOT * KTOT / 4;     // 16,777,216 float4s
    size_t idx    = (size_t)blockIdx.x * 256 + threadIdx.x;
    size_t stride = (size_t)gridDim.x * 256;
    const f32x4* w4 = (const f32x4*)w;
    float s = 0.0f;
    for (size_t c = idx; c < nchunk; c += stride) {
        f32x4 v = w4[c];
#pragma unroll
        for (int i = 0; i < 4; ++i) {
            float f = fminf(fmaxf(v[i], -2.0f), 2.0f);
            s += fabsf(f);
        }
    }
    red[threadIdx.x] = s;
    __syncthreads();
#pragma unroll
    for (int off = 128; off > 0; off >>= 1) {
        if ((int)threadIdx.x < off) red[threadIdx.x] += red[threadIdx.x + off];
        __syncthreads();
    }
    if (threadIdx.x == 0) partial[blockIdx.x] = red[0];
}

// ---- kernel 2: finalize gamma (single block, deterministic tree) -----------
__global__ __launch_bounds__(256) void k_gamma(const float* __restrict__ partial,
                                               float* __restrict__ gammap) {
    __shared__ float red[256];
    int t = threadIdx.x;
    float s = partial[t] + partial[t + 256] + partial[t + 512] + partial[t + 768];
    red[t] = s;
    __syncthreads();
#pragma unroll
    for (int off = 128; off > 0; off >>= 1) {
        if (t < off) red[t] += red[t + off];
        __syncthreads();
    }
    if (t == 0) {
        float g = red[0] / (float)((size_t)NTOT * KTOT);
        *gammap = fmaxf(g, 1e-4f);
    }
}

// ---- kernel 3: quantize weight -> ternary bf16 ({-1,0,1} exact) ------------
__global__ __launch_bounds__(256) void k_quant_w(const float* __restrict__ w,
                                                 const float* __restrict__ gammap,
                                                 unsigned short* __restrict__ wq) {
    const float inv_g = 1.0f / *gammap;
    size_t c = (size_t)blockIdx.x * 256 + threadIdx.x;
    f32x4 v = ((const f32x4*)w)[c];
    u16x4 q;
#pragma unroll
    for (int i = 0; i < 4; ++i) {
        float f = fminf(fmaxf(v[i], -2.0f), 2.0f);
        float t = fminf(fmaxf(rintf(f * inv_g), -1.0f), 1.0f);
        q[i] = f32_to_bf16_rne(t);
    }
    ((u16x4*)wq)[c] = q;
}

// ---- kernel 4: x f32 -> bf16 ----------------------------------------------
__global__ __launch_bounds__(256) void k_cvt_x(const float* __restrict__ x,
                                               unsigned short* __restrict__ xq) {
    size_t c = (size_t)blockIdx.x * 256 + threadIdx.x;
    f32x4 v = ((const f32x4*)x)[c];
    u16x4 q;
#pragma unroll
    for (int i = 0; i < 4; ++i) q[i] = f32_to_bf16_rne(v[i]);
    ((u16x4*)xq)[c] = q;
}

// ---- kernel 5: WMMA GEMM with 3-stage TDM pipeline -------------------------
// Block tile 128(M) x 128(N), K-step 32. 256 threads = 8 waves (2x4 wave grid),
// wave tile 64x32 = 4x2 WMMA tiles. LDS tiles stored [row][k]; TDM pads each
// 64B row with 16B (pad_interval=16 dwords, pad_amount=4 dwords) -> 80B stride
// so fragment ds_load_b128s are bank-conflict-free. Triple-buffered, prefetch
// distance 2: steady state waits TENSORcnt<=1 so one DMA is always in flight.
#define BM 128
#define BN 128
#define BK 32
#define LDSW 40     // row stride in bf16 elements (80 bytes)
#define STAGES 3

// Issue one TDM 2D tile load: tile BK x BM (x-major rows of 32 bf16) from a
// row-major [rows x KTOT] bf16 tensor into LDS at lds_off, padded to 80B rows.
// D# bitfields per CDNA5 ISA 8.3/8.4. Groups 2/3 zero (tile_dim2..4 unused).
__device__ __forceinline__ void tdm_load_tile(const unsigned short* gptr,
                                              unsigned int lds_off,
                                              unsigned int tensor_rows) {
    unsigned long long ga = (unsigned long long)(uintptr_t)gptr;
    tdm_g0_t g0;
    g0[0] = 1u;                                          // count=1, user D#
    g0[1] = lds_off;                                     // lds_addr
    g0[2] = (unsigned int)ga;                            // global_addr[31:0]
    g0[3] = ((unsigned int)(ga >> 32) & 0x01FFFFFFu)     // global_addr[56:32]
          | (2u << 30);                                  // type = 2 ("image")
    tdm_g1_t g1;
    g1[0] = (int)((1u << 16)        // data_size = 2 bytes (bf16)
                | (1u << 20)        // pad_enable
                | (3u << 22)        // pad_interval: 16 dwords (one 64B row)
                | (3u << 25));      // pad_amount:   4 dwords (16B) -> 80B rows
    g1[1] = (int)(((unsigned)KTOT & 0xFFFFu) << 16);     // tensor_dim0 lo16
    g1[2] = (int)((((unsigned)KTOT >> 16) & 0xFFFFu)     // tensor_dim0 hi16
                | ((tensor_rows & 0xFFFFu) << 16));      // tensor_dim1 lo16
    g1[3] = (int)(((tensor_rows >> 16) & 0xFFFFu)        // tensor_dim1 hi16
                | ((unsigned)BK << 16));                 // tile_dim0 = 32
    g1[4] = (int)(unsigned)BM;                           // tile_dim1=128, dim2=0
    g1[5] = (int)(unsigned)KTOT;                         // tensor_dim0_stride lo32
    g1[6] = 0;                                           // stride hi / dim1_stride lo
    g1[7] = 0;                                           // dim1_stride hi
    tdm_g2_t z = (tdm_g2_t)0;
#if __clang_major__ >= 23
    __builtin_amdgcn_tensor_load_to_lds(g0, g1, z, z, (tdm_g8_t)0, 0);
#else
    __builtin_amdgcn_tensor_load_to_lds(g0, g1, z, z, 0);
#endif
}

__global__ __launch_bounds__(256) void k_gemm(const unsigned short* __restrict__ xq,
                                              const unsigned short* __restrict__ wq,
                                              const float* __restrict__ bias,
                                              const float* __restrict__ gammap,
                                              float* __restrict__ out) {
    __shared__ __align__(16) unsigned short sA[STAGES][BM * LDSW];
    __shared__ __align__(16) unsigned short sB[STAGES][BN * LDSW];

    const int tid  = threadIdx.x;
    const int n0   = blockIdx.x * BN;
    const int m0   = blockIdx.y * BM;
    const float gamma = *gammap;

    const int lane = tid & 31;
    const int wm   = (tid >> 5) >> 2;   // 0..1  (M direction)
    const int wn   = (tid >> 5) & 3;    // 0..3  (N direction)
    const int lr   = lane & 15;         // row (A) / col (B) within fragment
    const int lg   = lane >> 4;         // K group
    const int kb   = lg * 8;            // low-chunk K offset (bf16 elems)

    // Scalar (SGPR) wave id so the TDM issue is guarded by a *scalar* branch:
    // TDM ignores EXEC, so an EXEC-masked path would issue 8 duplicate DMAs.
    const unsigned swave = __builtin_amdgcn_readfirstlane((unsigned)tid) >> 5;

    const unsigned ldsA0 = (unsigned)(uintptr_t)&sA[0][0];
    const unsigned ldsB0 = (unsigned)(uintptr_t)&sB[0][0];
    const unsigned bufBytes = BM * LDSW * 2;             // 10240 B per stage

    const unsigned short* gA = xq + (size_t)m0 * KTOT;   // A tile row origin
    const unsigned short* gB = wq + (size_t)n0 * KTOT;   // B tile row origin

    v8f acc[4][2] = {};

    // prologue: DMA tiles 0 and 1 into stages 0 and 1
    if (swave == 0) {
        tdm_load_tile(gA,      ldsA0,            (unsigned)MTOT);
        tdm_load_tile(gA + BK, ldsA0 + bufBytes, (unsigned)MTOT);
    }
    if (swave == 1) {
        tdm_load_tile(gB,      ldsB0,            (unsigned)NTOT);
        tdm_load_tile(gB + BK, ldsB0 + bufBytes, (unsigned)NTOT);
    }
    __builtin_amdgcn_s_wait_tensorcnt(1);   // tile 0 landed; tile 1 in flight
    __syncthreads();

    const int ITERS = KTOT / BK;     // 128
    for (int kt = 0; kt < ITERS; ++kt) {
        const int buf = kt % STAGES;

        if (kt + 2 < ITERS) {        // DMA tile kt+2 into stage (kt+2)%3
            const unsigned off = (unsigned)((kt + 2) % STAGES) * bufBytes;
            if (swave == 0) tdm_load_tile(gA + (size_t)(kt + 2) * BK, ldsA0 + off, (unsigned)MTOT);
            if (swave == 1) tdm_load_tile(gB + (size_t)(kt + 2) * BK, ldsB0 + off, (unsigned)NTOT);
        }

        // fragments (ISA 7.12.2 16-bit A layout; B mirrors with lane = N)
        v16bf afrag[4], bfrag[2];
#pragma unroll
        for (int i = 0; i < 4; ++i) {
            const int row = wm * 64 + i * 16 + lr;
            v8bf lo = *(const v8bf*)&sA[buf][row * LDSW + kb];
            v8bf hi = *(const v8bf*)&sA[buf][row * LDSW + kb + 16];
            afrag[i] = __builtin_shufflevector(lo, hi, 0,1,2,3,4,5,6,7,8,9,10,11,12,13,14,15);
        }
#pragma unroll
        for (int j = 0; j < 2; ++j) {
            const int row = wn * 32 + j * 16 + lr;
            v8bf lo = *(const v8bf*)&sB[buf][row * LDSW + kb];
            v8bf hi = *(const v8bf*)&sB[buf][row * LDSW + kb + 16];
            bfrag[j] = __builtin_shufflevector(lo, hi, 0,1,2,3,4,5,6,7,8,9,10,11,12,13,14,15);
        }

#pragma unroll
        for (int i = 0; i < 4; ++i)
#pragma unroll
            for (int j = 0; j < 2; ++j)
                acc[i][j] = __builtin_amdgcn_wmma_f32_16x16x32_bf16(
                    false, afrag[i], false, bfrag[j], (short)0, acc[i][j], false, false);

        if (kt + 1 < ITERS) {
            // Retire the oldest in-flight DMA (tile kt+1, TENSORcnt is in-order).
            // When no DMA was issued this iteration, drain fully instead.
            if (kt + 2 < ITERS) __builtin_amdgcn_s_wait_tensorcnt(1);
            else                __builtin_amdgcn_s_wait_tensorcnt(0);
            __syncthreads();                    // publish stage to all waves
        }
    }

    // epilogue: D layout -> lane: n = lr, rows = vgpr + lg*8; fuse gamma+bias.
    // out (512MB) is write-once: non-temporal stores keep it from evicting the
    // reused bf16 operands from the 192MB L2.
#pragma unroll
    for (int i = 0; i < 4; ++i) {
        const int mbase = m0 + wm * 64 + i * 16 + lg * 8;
#pragma unroll
        for (int j = 0; j < 2; ++j) {
            const int n = n0 + wn * 32 + j * 16 + lr;
            const float b = bias[n];
#pragma unroll
            for (int r = 0; r < 8; ++r)
                __builtin_nontemporal_store(acc[i][j][r] * gamma + b,
                                            &out[(size_t)(mbase + r) * NTOT + n]);
        }
    }
}

// ---------------------------------------------------------------------------
extern "C" void kernel_launch(void* const* d_in, const int* in_sizes, int n_in,
                              void* d_out, int out_size, void* d_ws, size_t ws_size,
                              hipStream_t stream) {
    const float* x    = (const float*)d_in[0];   // [4,2048,4096] f32
    const float* w    = (const float*)d_in[1];   // [16384,4096] f32
    const float* bias = (const float*)d_in[2];   // [16384] f32
    float*       out  = (float*)d_out;           // [4,2048,16384] f32

    // workspace layout (needs ~201 MB):
    //   [0,4096)          : 1024 partial sums
    //   [4096,4100)       : gamma
    //   [8192, +64MiB)    : x as bf16
    //   [.., +128MiB)     : w quantized as ternary bf16
    char* ws = (char*)d_ws;
    float*          partial = (float*)ws;
    float*          gammap  = (float*)(ws + 4096);
    unsigned short* xq      = (unsigned short*)(ws + 8192);
    unsigned short* wqp     = (unsigned short*)(ws + 8192 + (size_t)MTOT * KTOT * 2);

    k_abs_partial<<<1024, 256, 0, stream>>>(w, partial);
    k_gamma<<<1, 256, 0, stream>>>(partial, gammap);
    k_quant_w<<<(int)(((size_t)NTOT * KTOT / 4) / 256), 256, 0, stream>>>(w, gammap, wqp);
    k_cvt_x<<<(int)(((size_t)MTOT * KTOT / 4) / 256), 256, 0, stream>>>(x, xq);

    dim3 grid(NTOT / BN, MTOT / BM);   // (128, 64)
    k_gemm<<<grid, 256, 0, stream>>>(xq, wqp, bias, gammap, out);
}
